// SingleHeadConv_54400055771646
// MI455X (gfx1250) — matrix operand order repaired and checked by reading, not compiled, not run on hardware
//
#include <hip/hip_runtime.h>
#include <hip/hip_bf16.h>

// ---------------------------------------------------------------------------
// CDNA5 WMMA types (wave32): v_wmma_f32_16x16x32_bf16
// ---------------------------------------------------------------------------
typedef __attribute__((ext_vector_type(16))) __bf16 bfx16;
typedef __attribute__((ext_vector_type(8)))  __bf16 bfx8;
typedef __attribute__((ext_vector_type(4)))  __bf16 bfx4;
typedef __attribute__((ext_vector_type(8)))  float  fx8;

__device__ __forceinline__ float silu_f(float v) {
    return v / (1.0f + __expf(-v));
}

__device__ __forceinline__ fx8 wmma_bf16(bfx16 a, bfx16 b, fx8 c) {
    return __builtin_amdgcn_wmma_f32_16x16x32_bf16(
        false, a, false, b, (short)0, c, false, false);
}

// ---------------------------------------------------------------------------
// Fragment builders (contiguous k = 16*(lane/16) + s, same order in A and B).
// ---------------------------------------------------------------------------
__device__ __forceinline__ bfx16 frag_a_lds(const __bf16* base, int lane, int stride) {
    const int m = lane & 15, hi = (lane >> 4) & 1;
    const __bf16* p = base + m * stride;
    bfx8 lo = *(const bfx8*)(p + (hi ? 16 : 0));
    bfx8 hh = *(const bfx8*)(p + (hi ? 24 : 8));
    return __builtin_shufflevector(lo, hh, 0,1,2,3,4,5,6,7,8,9,10,11,12,13,14,15);
}

__device__ __forceinline__ bfx16 frag_b_lds(const __bf16* base, int lane, int stride) {
    const int n = lane & 15, kk = ((lane >> 4) & 1) * 16;
    const __bf16* p = base + n * stride + kk;
    bfx8 lo = *(const bfx8*)(p);
    bfx8 hh = *(const bfx8*)(p + 8);
    return __builtin_shufflevector(lo, hh, 0,1,2,3,4,5,6,7,8,9,10,11,12,13,14,15);
}

template <typename G>
__device__ __forceinline__ bfx16 make_frag_a(int lane, G get) {
    bfx16 v;
    const int m  = lane & 15;
    const int hi = (lane >> 4) & 1;
#pragma unroll
    for (int j = 0; j < 16; ++j) {
        const int k = j + (hi ? 16 : 0);
        v[j] = (__bf16)get(m, k);
    }
    return v;
}
template <typename G>
__device__ __forceinline__ bfx16 make_frag_b(int lane, G get) {
    bfx16 v;
    const int n  = lane & 15;
    const int k0 = ((lane >> 4) & 1) * 16;
#pragma unroll
    for (int j = 0; j < 16; ++j) v[j] = (__bf16)get(k0 + j, n);
    return v;
}

// ---------------------------------------------------------------------------
#define F_IN  256
#define C_CH  32
#define A_SH  9
#define WNUM  288
#define B_RAD 8
#define H_MLP 64
#define S_SP  4
#define F_OUT 256

#define ASTR 40   // LDS slab stride in bf16 (80B: 16B aligned, bank-spread)
#define BSTR 40

// ---------------------------------------------------------------------------
// Tiled-GEMM skeleton: 256 threads (8 waves), block tile MB x NB, K step 32.
// Software pipelined: global gathers land in registers BEFORE the barrier
// (overlapping the previous round's WMMA), stores to LDS after it.
// Waves own MSUB x NSUB/..: NACC accumulators per wave when MB*NB/256 > 8.
// ---------------------------------------------------------------------------
template <int MB, int NB, int NK, typename AG, typename BG, typename EP>
__device__ __forceinline__ void gemm_tile(AG aget, BG bget, EP epi) {
    constexpr int MSUB  = MB / 16;
    constexpr int NSUB  = NB / 16;
    constexpr int NACC  = (MSUB * NSUB) / 8;     // tiles per wave
    constexpr int APASS = (MB * 32) / 1024;
    constexpr int BPASS = (NB * 32) / 1024;

    __shared__ __align__(16) __bf16 sA[MB * ASTR];
    __shared__ __align__(16) __bf16 sB[NB * BSTR];
    const int tid = threadIdx.x, lane = tid & 31, wid = tid >> 5;
    const int msub = wid % MSUB;
    const int ngrp = wid / MSUB;

    fx8 acc[NACC];
#pragma unroll
    for (int i = 0; i < NACC; ++i) acc[i] = (fx8){};

    for (int kk = 0; kk < NK; ++kk) {
        const int k0 = kk * 32;
        // ---- gather phase: all global loads into registers ----
        float areg[APASS][4], breg[BPASS][4];
#pragma unroll
        for (int p = 0; p < APASS; ++p) {
            const int idx = p * 1024 + tid * 4;
            const int r = idx >> 5, kc = idx & 31;
#pragma unroll
            for (int q = 0; q < 4; ++q) areg[p][q] = aget(r, k0 + kc + q);
        }
#pragma unroll
        for (int p = 0; p < BPASS; ++p) {
            const int idx = p * 1024 + tid * 4;
            const int kB = idx / NB, nB = idx % NB;
#pragma unroll
            for (int q = 0; q < 4; ++q) breg[p][q] = bget(k0 + kB, nB + q);
        }
        __syncthreads();   // previous round's fragment reads complete
        // ---- store phase: cvt + LDS ----
#pragma unroll
        for (int p = 0; p < APASS; ++p) {
            const int idx = p * 1024 + tid * 4;
            const int r = idx >> 5, kc = idx & 31;
            bfx4 pk;
#pragma unroll
            for (int q = 0; q < 4; ++q) pk[q] = (__bf16)areg[p][q];
            *(bfx4*)(sA + r * ASTR + kc) = pk;
        }
#pragma unroll
        for (int p = 0; p < BPASS; ++p) {
            const int idx = p * 1024 + tid * 4;
            const int kB = idx / NB, nB = idx % NB;
#pragma unroll
            for (int q = 0; q < 4; ++q)
                sB[(nB + q) * BSTR + kB] = (__bf16)breg[p][q];
        }
        __syncthreads();
        // ---- math phase ----
        bfx16 a = frag_a_lds(sA + msub * 16 * ASTR, lane, ASTR);
#pragma unroll
        for (int i = 0; i < NACC; ++i) {
            bfx16 b = frag_b_lds(sB + (ngrp * NACC + i) * 16 * BSTR, lane, BSTR);
            acc[i] = wmma_bf16(a, b, acc[i]);
        }
    }
#pragma unroll
    for (int i = 0; i < NACC; ++i) epi(acc[i], msub, ngrp * NACC + i, lane);
}

// ---------------------------------------------------------------------------
__global__ void k_zero(float* __restrict__ p, long n) {
    long i = (long)blockIdx.x * blockDim.x + threadIdx.x;
    long stride = (long)gridDim.x * blockDim.x;
    for (; i < n; i += stride) p[i] = 0.0f;
}

// linear_1: h[N,32] = (x @ W1) * 0.25 ; block tile 64x32, K=256
__global__ __launch_bounds__(256) void k_linear1(const float* __restrict__ x,
                                                 const float* __restrict__ W1,
                                                 float* __restrict__ h, int N) {
    const int m0 = blockIdx.x * 64;
    gemm_tile<64, 32, F_IN / 32>(
        [&](int r, int k) -> float {
            const int row = m0 + r;
            const int rc = row < N ? row : N - 1;
            const float v = x[(size_t)rc * F_IN + k];
            return row < N ? v : 0.0f;
        },
        [&](int k, int n) -> float { return W1[(size_t)k * C_CH + n]; },
        [&](fx8 acc, int msub, int nsub, int lane) {
            const int n = lane & 15, hi = (lane >> 4) & 1;
#pragma unroll
            for (int r = 0; r < 8; ++r) {
                const int row = m0 + msub * 16 + r + hi * 8;
                if (row < N) h[(size_t)row * C_CH + nsub * 16 + n] = acc[r] * 0.25f;
            }
        });
}

// self-connection: out = einsum('nf,ns,fso->no'); A[m,f*4+s]=x*attrs, K=1024
__global__ __launch_bounds__(256) void k_selfconn(const float* __restrict__ x,
                                                  const float* __restrict__ attrs,
                                                  const float* __restrict__ Wsc,
                                                  float* __restrict__ out, int N) {
    const int m0 = blockIdx.x * 64;
    const int nb = blockIdx.y * 64;
    gemm_tile<64, 64, (F_IN * S_SP) / 32>(
        [&](int r, int k) -> float {
            const int row = m0 + r;
            const int rc = row < N ? row : N - 1;
            const float v = x[(size_t)rc * F_IN + (k >> 2)] *
                            attrs[(size_t)rc * S_SP + (k & 3)];
            return row < N ? v : 0.0f;
        },
        [&](int k, int n) -> float { return Wsc[(size_t)k * F_OUT + nb + n]; },
        [&](fx8 acc, int msub, int nsub, int lane) {
            const int n = lane & 15, hi = (lane >> 4) & 1;
#pragma unroll
            for (int r = 0; r < 8; ++r) {
                const int row = m0 + msub * 16 + r + hi * 8;
                if (row < N) out[(size_t)row * F_OUT + nb + nsub * 16 + n] = acc[r];
            }
        });
}

// linear_2 + silu + add self-connection already in out
__global__ __launch_bounds__(256) void k_linear2(const float* __restrict__ agg,
                                                 const float* __restrict__ W2,
                                                 float* __restrict__ out, int N) {
    const int m0 = blockIdx.x * 64;
    const int nb = blockIdx.y * 64;
    gemm_tile<64, 64, WNUM / 32>(
        [&](int r, int k) -> float {
            const int row = m0 + r;
            const int rc = row < N ? row : N - 1;
            const float v = agg[(size_t)rc * WNUM + k];
            return row < N ? v : 0.0f;
        },
        [&](int k, int n) -> float { return W2[(size_t)k * F_OUT + nb + n]; },
        [&](fx8 acc, int msub, int nsub, int lane) {
            const int n = lane & 15, hi = (lane >> 4) & 1;
#pragma unroll
            for (int r = 0; r < 8; ++r) {
                const int row = m0 + msub * 16 + r + hi * 8;
                if (row < N) {
                    const size_t idx = (size_t)row * F_OUT + nb + nsub * 16 + n;
                    out[idx] = silu_f(acc[r]) + out[idx];
                }
            }
        });
}

// ---------------------------------------------------------------------------
// Fused edge kernel: persistent blocks, grid-stride over 64-edge tiles.
// Weights transposed->LDS (bf16) once per block; 3-layer MLP via WMMA;
// TP + atomic scatter fused on each 16x16 tile of the layer-3 output.
// 128 threads = 4 waves, 16 edges/wave.  LDS = 64,640 B.
// ---------------------------------------------------------------------------
#define EW 4
#define W2STR 72   // 144B rows: 16B aligned, conflict-free for 16 lanes
#define W3STR 72
#define SLSTR 64   // slab row stride (128B, 16B aligned)

__global__ __launch_bounds__(128) void k_edge(const float* __restrict__ emb,
                                              const float* __restrict__ shat,
                                              const int*   __restrict__ eidx,
                                              const float* __restrict__ h,
                                              const float* __restrict__ Wm1,
                                              const float* __restrict__ Wm2,
                                              const float* __restrict__ Wm3,
                                              float* __restrict__ agg, int E) {
    __shared__ __align__(16) __bf16 sW2t[H_MLP * W2STR];  // [n][k]   9216 B
    __shared__ __align__(16) __bf16 sW3t[WNUM * W3STR];   // [n][k]  41472 B
    __shared__ __align__(16) __bf16 slab[EW][16 * SLSTR]; //          8192 B
    __shared__ __bf16 sH[EW][16 * C_CH];                  //          4096 B
    __shared__ __bf16 sSH[EW][16 * A_SH];                 //          1152 B
    __shared__ int    sDst[EW][16];                       //           256 B
    __shared__ int    sSrc[EW][16];                       //           256 B

    const int tid = threadIdx.x, lane = tid & 31, w = tid >> 5;

    // stage weights transposed (amortized over all tiles of this block)
    for (int i = tid; i < H_MLP * H_MLP; i += 128) {
        const int k = i >> 6, n = i & 63;
        sW2t[n * W2STR + k] = (__bf16)Wm2[i];
    }
    for (int i = tid; i < H_MLP * WNUM; i += 128) {
        const int k = i / WNUM, n = i % WNUM;
        sW3t[n * W3STR + k] = (__bf16)Wm3[i];
    }

    const int nTiles = (E + EW * 16 - 1) / (EW * 16);
    for (int t = blockIdx.x; t < nTiles; t += gridDim.x) {
        const int e0 = t * (EW * 16) + w * 16;
        __syncthreads();   // prior-tile LDS reads complete (also covers weights)

        if (lane < 16) {
            const int e = e0 + lane;
            const int ec = e < E ? e : E - 1;
            sDst[w][lane] = e < E ? eidx[ec] : -1;
            sSrc[w][lane] = e < E ? eidx[E + ec] : 0;
        }
        for (int i = lane; i < 16 * A_SH; i += 32) {
            const int e = e0 + i / A_SH;
            const int ec = e < E ? e : E - 1;
            const float v = shat[(size_t)ec * A_SH + i % A_SH];
            sSH[w][i] = (__bf16)(e < E ? v : 0.0f);
        }
        __syncthreads();   // sSrc visible
        for (int it = 0; it < 16; ++it) {     // coalesced rows of h[src]
            sH[w][it * C_CH + lane] = (__bf16)h[(size_t)sSrc[w][it] * C_CH + lane];
        }

        // ---- layer 1: [16x8]@[8x64], K padded to 32, branch-free masks ----
        bfx16 a1 = make_frag_a(lane, [&](int m, int k) -> float {
            const int e = e0 + m;
            const int ec = e < E ? e : E - 1;
            const float v = emb[(size_t)ec * B_RAD + (k & 7)];
            return (k < B_RAD && e < E) ? v : 0.0f;
        });
        fx8 o1[4];
#pragma unroll
        for (int nt = 0; nt < 4; ++nt) {
            bfx16 b = make_frag_b(lane, [&](int k, int n) -> float {
                const float v = Wm1[(size_t)(k & 7) * H_MLP + nt * 16 + n];
                return k < B_RAD ? v : 0.0f;
            });
            fx8 z = {};
            o1[nt] = wmma_bf16(a1, b, z);
        }
        __syncthreads();
        {
            const int n = lane & 15, hi = (lane >> 4) & 1;
#pragma unroll
            for (int nt = 0; nt < 4; ++nt)
#pragma unroll
                for (int r = 0; r < 8; ++r)
                    slab[w][(r + hi * 8) * SLSTR + nt * 16 + n] =
                        (__bf16)silu_f(o1[nt][r]);
        }
        __syncthreads();

        // ---- layer 2: [16x64]@[64x64] ------------------------------------
        bfx16 aLo = frag_a_lds(slab[w], lane, SLSTR);
        bfx16 aHi = frag_a_lds(slab[w] + 32, lane, SLSTR);
        __syncthreads();   // frag reads done before slab overwrite
        fx8 o2[4];
#pragma unroll
        for (int nt = 0; nt < 4; ++nt) {
            fx8 acc = {};
            acc = wmma_bf16(aLo, frag_b_lds(sW2t + nt * 16 * W2STR, lane, W2STR), acc);
            acc = wmma_bf16(aHi, frag_b_lds(sW2t + nt * 16 * W2STR + 32, lane, W2STR), acc);
            o2[nt] = acc;
        }
        {
            const int n = lane & 15, hi = (lane >> 4) & 1;
#pragma unroll
            for (int nt = 0; nt < 4; ++nt)
#pragma unroll
                for (int r = 0; r < 8; ++r)
                    slab[w][(r + hi * 8) * SLSTR + nt * 16 + n] =
                        (__bf16)silu_f(o2[nt][r]);
        }
        __syncthreads();

        // ---- layer 3: [16x64]@[64x288] + TP + atomic scatter -------------
        bfx16 cLo = frag_a_lds(slab[w], lane, SLSTR);
        bfx16 cHi = frag_a_lds(slab[w] + 32, lane, SLSTR);
        const int n = lane & 15, hi = (lane >> 4) & 1;
#pragma unroll 2
        for (int nt = 0; nt < WNUM / 16; ++nt) {
            fx8 acc = {};
            acc = wmma_bf16(cLo, frag_b_lds(sW3t + nt * 16 * W3STR, lane, W3STR), acc);
            acc = wmma_bf16(cHi, frag_b_lds(sW3t + nt * 16 * W3STR + 32, lane, W3STR), acc);

            const int col = nt * 16 + n;          // c = col/9, a = col%9
            const int c = col / A_SH, a = col % A_SH;
#pragma unroll
            for (int r = 0; r < 8; ++r) {
                const int m = r + hi * 8;
                const int d = sDst[w][m];
                if (d >= 0) {
                    const float msg = acc[r] * (float)sH[w][m * C_CH + c] *
                                      (float)sSH[w][m * A_SH + a];
                    atomicAdd(&agg[(size_t)d * WNUM + col], msg);
                }
            }
        }
    }
}

// ---------------------------------------------------------------------------
extern "C" void kernel_launch(void* const* d_in, const int* in_sizes, int n_in,
                              void* d_out, int out_size, void* d_ws, size_t ws_size,
                              hipStream_t stream) {
    const float* x     = (const float*)d_in[0];
    const float* attrs = (const float*)d_in[1];
    const float* emb   = (const float*)d_in[2];
    const float* shat  = (const float*)d_in[3];
    const int*   eidx  = (const int*)d_in[4];
    const float* W1    = (const float*)d_in[5];
    const float* Wm1   = (const float*)d_in[6];
    const float* Wm2   = (const float*)d_in[7];
    const float* Wm3   = (const float*)d_in[8];
    const float* W2    = (const float*)d_in[9];
    const float* Wsc   = (const float*)d_in[10];
    float* out = (float*)d_out;

    const int N = in_sizes[0] / F_IN;   // 25000
    const int E = in_sizes[3] / A_SH;   // 400000

    float* h   = (float*)d_ws;                 // N*32 f32
    float* agg = h + (size_t)N * C_CH;         // N*288 f32

    k_zero<<<1024, 256, 0, stream>>>(agg, (long)N * WNUM);

    k_linear1<<<dim3((N + 63) / 64), 256, 0, stream>>>(x, W1, h, N);
    k_selfconn<<<dim3((N + 63) / 64, F_OUT / 64), 256, 0, stream>>>(x, attrs, Wsc, out, N);

    const int nTiles = (E + EW * 16 - 1) / (EW * 16);
    const int egrid = nTiles < 2048 ? nTiles : 2048;
    k_edge<<<egrid, 128, 0, stream>>>(emb, shat, eidx, h, Wm1, Wm2, Wm3, agg, E);

    k_linear2<<<dim3((N + 63) / 64, F_OUT / 64), 256, 0, stream>>>(agg, W2, out, N);
}
